// BIMPM_85864986182108
// MI455X (gfx1250) — matrix operand order repaired and checked by reading, not compiled, not run on hardware
//
#include <hip/hip_runtime.h>

// ---------------------------------------------------------------------------
// BiMPM forward for MI455X (gfx1250): bf16 WMMA GEMMs + persistent LSTM scans
// B-operands pre-packed into WMMA fragment order (contiguous 32B per lane)
// ---------------------------------------------------------------------------

typedef __bf16 bf16_t;
typedef __attribute__((ext_vector_type(16))) __bf16 v16bf;
typedef __attribute__((ext_vector_type(8)))  float  v8f;

namespace {
constexpr int Bq = 64;      // batch
constexpr int Sq = 128;     // seq len
constexpr int Dq = 300;     // embed dim
constexpr int Hq = 128;     // hidden
constexpr int Lq = 20;      // perspectives
constexpr int Cq = 3;       // classes
constexpr int Dp = 320;     // D padded to multiple of 32 (WMMA K step)
constexpr int G4 = 512;     // 4*H gate width
constexpr float EPSV = 1e-8f;
}

// ---------------- WMMA fragment helpers (wave32, 16x16x32 bf16) -------------
// A 16x32 bf16 layout: lane m=lane&15, half=lane>>4;
//  vgpr j -> K = half*16 + 2j, 2j+1
//  (one contiguous 32-byte run per lane)
// B 32x16 bf16: lane n=lane&15, half; vgpr j -> K = half*16 + 2j, 2j+1
// C/D f32 16x16: vgpr r -> M = r + 8*half, N = lane&15

__device__ inline v8f zero8() {
  v8f z;
#pragma unroll
  for (int i = 0; i < 8; ++i) z[i] = 0.0f;
  return z;
}

__device__ inline int a_kpair(int j, int half) {
  return half * 16 + 2 * j;
}

__device__ inline v16bf load_a_bf16(const bf16_t* A, int lda, int lane) {
  int m = lane & 15, half = lane >> 4;
  v16bf r;
#pragma unroll
  for (int j = 0; j < 8; ++j) {
    int k = a_kpair(j, half);
    const bf16_t* p = A + m * lda + k;
    r[2 * j]     = p[0];
    r[2 * j + 1] = p[1];
  }
  return r;
}

// B operand from row-major [n][k] source: element(k,n) = Bm[n*ld + k]
// (contiguous 16-element run per lane)
__device__ inline v16bf load_b_tr(const bf16_t* Bm, int ld, int lane) {
  int n = lane & 15, half = lane >> 4;
  v16bf r;
#pragma unroll
  for (int j = 0; j < 8; ++j) {
    int k = half * 16 + 2 * j;
    r[2 * j]     = Bm[n * ld + k];
    r[2 * j + 1] = Bm[n * ld + k + 1];
  }
  return r;
}

// B operand from fragment-packed storage: block base, then lane*16 contiguous
__device__ inline v16bf load_b_packed(const bf16_t* Bp, int lane) {
  return *(const v16bf*)(Bp + lane * 16);
}

__device__ inline v8f wmma_bf16(v16bf a, v16bf b, v8f c) {
  return __builtin_amdgcn_wmma_f32_16x16x32_bf16(false, a, false, b, (short)0, c,
                                                 false, false);
}

__device__ inline float sigf(float x) { return 1.0f / (1.0f + expf(-x)); }

// order-preserving float <-> uint (for LDS atomic max on possibly-negative cos)
__device__ inline unsigned fenc(float f) {
  unsigned u = __float_as_uint(f);
  return (u >> 31) ? ~u : (u | 0x80000000u);
}
__device__ inline float fdec(unsigned e) {
  unsigned u = (e >> 31) ? (e & 0x7fffffffu) : ~e;
  return __uint_as_float(u);
}

// ---------------------------------------------------------------------------
// Small prep kernels
// ---------------------------------------------------------------------------

// src (N,K) row-major f32 -> WMMA-B fragment-packed bf16, zero-padded to Kp.
// Block (kb,nb) of (32k x 16n): 32 lanes x 16 contiguous bf16 per lane.
// lane l: n = l&15, half = l>>4 holds k = kb*32 + half*16 + e (e=0..15)
__global__ void k_wpack(const float* __restrict__ src, bf16_t* __restrict__ dst,
                        int N, int K, int Kp) {
  long idx = (long)blockIdx.x * blockDim.x + threadIdx.x;
  if (idx >= (long)Kp * N) return;
  int e = (int)(idx & 15);
  int l = (int)((idx >> 4) & 31);
  long blk = idx >> 9;
  int ntn = N >> 4;
  int kb = (int)(blk / ntn), nb = (int)(blk % ntn);
  int n = l & 15, half = l >> 4;
  int k = kb * 32 + half * 16 + e;
  int col = nb * 16 + n;
  dst[idx] = (k < K) ? (bf16_t)src[(long)col * K + k] : (bf16_t)0.0f;
}

__global__ void k_cvt_bf16(const float* __restrict__ src, bf16_t* __restrict__ dst,
                           int n) {
  int i = blockIdx.x * blockDim.x + threadIdx.x;
  if (i < n) dst[i] = (bf16_t)src[i];
}

// embedding gather: out[(b*S+s), 0..Dp) = bf16(emb[tok]) padded
__global__ void k_embed(const float* __restrict__ emb, const int* __restrict__ tok,
                        bf16_t* __restrict__ out) {
  long bs = blockIdx.x;
  long t = tok[bs];
  const float* src = emb + t * Dq;
  bf16_t* dst = out + bs * Dp;
  for (int k = threadIdx.x; k < Dp; k += blockDim.x)
    dst[k] = (k < Dq) ? (bf16_t)src[k] : (bf16_t)0.0f;
}

// ---------------------------------------------------------------------------
// Generic WMMA GEMM: C[M,N] = A[M,K](bf16,rm) * B(fragment-packed) + bias
// one 16x16 tile per wave, 8 waves per block
// ---------------------------------------------------------------------------
__global__ __launch_bounds__(256)
void k_gemm(const bf16_t* __restrict__ A, const bf16_t* __restrict__ Bp,
            const float* __restrict__ bias, float* __restrict__ C,
            int M, int N, int K) {
  int wave = threadIdx.x >> 5, lane = threadIdx.x & 31;
  int tile = blockIdx.x * 8 + wave;
  int ntn = N >> 4;
  int mt = tile / ntn, nt = tile % ntn;
  if (mt * 16 >= M) return;
  v8f acc = zero8();
  const bf16_t* Ab = A + (long)mt * 16 * K;
  for (int k0 = 0; k0 < K; k0 += 32) {
    const bf16_t* bp = Bp + ((((long)(k0 >> 5)) * ntn + nt) << 9);
    acc = wmma_bf16(load_a_bf16(Ab + k0, K, lane), load_b_packed(bp, lane), acc);
  }
  int n = lane & 15, half = lane >> 4;
#pragma unroll
  for (int r = 0; r < 8; ++r) {
    int row = mt * 16 + r + 8 * half;
    int col = nt * 16 + n;
    C[(long)row * N + col] = acc[r] + (bias ? bias[col] : 0.0f);
  }
}

// ---------------------------------------------------------------------------
// Persistent LSTM scan: one block owns 16 batch rows for the full sequence.
// whh (fragment-packed bf16, 128 KB) resident in LDS; h (bf16), c (f32) in LDS.
// ---------------------------------------------------------------------------
__global__ __launch_bounds__(256)
void k_lstm(const float* __restrict__ xg,   // (B,S,512)
            const bf16_t* __restrict__ whh, // fragment-packed (K=128,N=512)
            float* __restrict__ hs,         // out base
            int out_stride, int out_off, int reverse) {
  const int BR = 16;
  int b0 = blockIdx.x * BR;
  extern __shared__ char smem[];
  bf16_t* s_whh = (bf16_t*)smem;                         // 128*512  (128 KB)
  float*  s_g   = (float*)(smem + 128 * 512 * 2);        // 16*512   (32 KB)
  float*  s_c   = s_g + 16 * 512;                        // 16*128   (8 KB)
  bf16_t* s_h   = (bf16_t*)(s_c + 16 * 128);             // 16*128   (4 KB)
  int tid = threadIdx.x, wave = tid >> 5, lane = tid & 31;

  for (int i = tid; i < 128 * 512; i += 256) s_whh[i] = whh[i];
  for (int i = tid; i < 16 * 128; i += 256) { s_c[i] = 0.0f; s_h[i] = (bf16_t)0.0f; }
  __syncthreads();

  const int ntn = 32; // 512/16
  for (int step = 0; step < Sq; ++step) {
    int s = reverse ? (Sq - 1 - step) : step;
    // g = h_prev @ whh^T  (16 x 512), 32 tiles over 8 waves
#pragma unroll
    for (int i = 0; i < 4; ++i) {
      int nt = wave * 4 + i;
      v8f acc = zero8();
#pragma unroll
      for (int k0 = 0; k0 < 128; k0 += 32)
        acc = wmma_bf16(load_a_bf16(s_h + k0, 128, lane),
                        load_b_packed(s_whh + (((k0 >> 5) * ntn + nt) << 9), lane),
                        acc);
      int n = lane & 15, half = lane >> 4;
#pragma unroll
      for (int r = 0; r < 8; ++r) {
        int row = r + 8 * half;
        int col = nt * 16 + n;
        s_g[row * 512 + col] =
            acc[r] + xg[((long)(b0 + row) * Sq + s) * G4 + col];
      }
    }
    __syncthreads();
    // gate nonlinearity for 16*128 units (torch order i,f,g,o)
    for (int u = tid; u < 16 * 128; u += 256) {
      int row = u >> 7, hcol = u & 127;
      const float* gr = s_g + row * 512;
      float gi = gr[hcol], gf = gr[128 + hcol], gg = gr[256 + hcol],
            go = gr[384 + hcol];
      float c = sigf(gf) * s_c[u] + sigf(gi) * tanhf(gg);
      float h = sigf(go) * tanhf(c);
      s_c[u] = c;
      s_h[u] = (bf16_t)h;
      hs[((long)(b0 + row) * Sq + s) * out_stride + out_off + hcol] = h;
    }
    __syncthreads();
  }
}

// ---------------------------------------------------------------------------
// Maxpool matching: one block per (b,l,dir). Cosine matrix 128x128 computed
// with WMMA from LDS-resident w-scaled tiles; max-reduced in LDS (never hits
// HBM). Writes aggr channels [40+dir*20 .. ) for both p1 and p2 sides.
// ---------------------------------------------------------------------------
__global__ __launch_bounds__(256)
void k_maxpool(const float* __restrict__ c1, const float* __restrict__ c2,
               const float* __restrict__ w3, const float* __restrict__ w4,
               float* __restrict__ aggr1, float* __restrict__ aggr2) {
  int id = blockIdx.x;
  int dir = id & 1; id >>= 1;
  int l = id % Lq;
  int b = id / Lq;
  int hoff = dir * Hq;
  const float* w = dir ? w4 : w3;

  extern __shared__ char smem[];
  bf16_t* sA = (bf16_t*)smem;                 // 128*128 bf16
  bf16_t* sB = sA + 128 * 128;                // 128*128 bf16
  float* sn1 = (float*)(sB + 128 * 128);      // 128
  float* sn2 = sn1 + 128;                     // 128
  unsigned* sm1 = (unsigned*)(sn2 + 128);     // 128
  unsigned* sm2 = sm1 + 128;                  // 128
  float* sW = (float*)(sm2 + 128);            // 128

  int tid = threadIdx.x, wave = tid >> 5, lane = tid & 31;
  if (tid < 128) sW[tid] = w[tid * Lq + l];
  __syncthreads();

  if (tid < 128) {
    int s = tid; float acc = 0.0f;
    const float* row = c1 + ((long)b * Sq + s) * (2 * Hq) + hoff;
    for (int h = 0; h < 128; ++h) {
      float v = row[h] * sW[h];
      sA[s * 128 + h] = (bf16_t)v;
      acc += v * v;
    }
    sn1[s] = sqrtf(acc);
    sm1[s] = 0u;
  } else {
    int t = tid - 128; float acc = 0.0f;
    const float* row = c2 + ((long)b * Sq + t) * (2 * Hq) + hoff;
    for (int h = 0; h < 128; ++h) {
      float v = row[h] * sW[h];
      sB[t * 128 + h] = (bf16_t)v;
      acc += v * v;
    }
    sn2[t] = sqrtf(acc);
    sm2[t] = 0u;
  }
  __syncthreads();

  // prod[s,t] = sum_h A[s,h]*B[t,h] ; 64 tiles over 8 waves
#pragma unroll
  for (int i = 0; i < 8; ++i) {
    int tile = wave * 8 + i;
    int mt = tile >> 3, nt = tile & 7;
    v8f acc = zero8();
#pragma unroll
    for (int k0 = 0; k0 < 128; k0 += 32)
      acc = wmma_bf16(load_a_bf16(sA + mt * 16 * 128 + k0, 128, lane),
                      load_b_tr(sB + (nt * 16) * 128 + k0, 128, lane), acc);
    int n = lane & 15, half = lane >> 4;
#pragma unroll
    for (int r = 0; r < 8; ++r) {
      int s = mt * 16 + r + 8 * half, t = nt * 16 + n;
      float deno = sn1[s] * sn2[t];
      float cosv = acc[r] / (deno > EPSV ? deno : EPSV);
      unsigned e = fenc(cosv);
      atomicMax(&sm1[s], e);
      atomicMax(&sm2[t], e);
    }
  }
  __syncthreads();
  int off = 2 * Lq + dir * Lq + l; // maxm block starts at channel 40
  if (tid < 128)
    aggr1[((long)b * Sq + tid) * (8 * Lq) + off] = fdec(sm1[tid]);
  else
    aggr2[((long)b * Sq + (tid - 128)) * (8 * Lq) + off] = fdec(sm2[tid - 128]);
}

// ---------------------------------------------------------------------------
// Attentive matching core: one block per (b,dir). Computes alpha (cosine),
// row-sums + argmax both ways, h_mat via WMMA, writes resultant & out_mat.
// Context tiles staged in both orientations; alpha stored bf16 both ways so
// every WMMA operand is a contiguous LDS run.
// ---------------------------------------------------------------------------
__global__ __launch_bounds__(256)
void k_attentive(const float* __restrict__ c1, const float* __restrict__ c2,
                 float* __restrict__ res1, float* __restrict__ res2,
                 float* __restrict__ om1, float* __restrict__ om2) {
  int id = blockIdx.x;
  int dir = id & 1;
  int b = id >> 1;
  int hoff = dir * Hq;
  const long BSH = (long)Bq * Sq * Hq;
  float* r1 = res1 + (long)dir * BSH;
  float* r2 = res2 + (long)dir * BSH;
  float* o1 = om1 + (long)dir * BSH;
  float* o2 = om2 + (long)dir * BSH;

  extern __shared__ char smem[];
  bf16_t* sA   = (bf16_t*)smem;        // [s][h] 128*128
  bf16_t* sB   = sA + 128 * 128;       // [t][h]
  bf16_t* sAT  = sB + 128 * 128;       // [h][s]
  bf16_t* sBT  = sAT + 128 * 128;      // [h][t]
  bf16_t* aBF  = sBT + 128 * 128;      // alpha  [s][t]
  bf16_t* aTBF = aBF + 128 * 128;      // alpha^T[t][s]
  float* sn1 = (float*)(aTBF + 128 * 128);
  float* sn2 = sn1 + 128;
  float* rs1 = sn2 + 128;
  float* rs2 = rs1 + 128;
  int* idx1 = (int*)(rs2 + 128);
  int* idx2 = idx1 + 128;

  int tid = threadIdx.x, wave = tid >> 5, lane = tid & 31;
  if (tid < 128) {
    int s = tid; float acc = 0.0f;
    const float* row = c1 + ((long)b * Sq + s) * (2 * Hq) + hoff;
    for (int h = 0; h < 128; ++h) {
      float v = row[h];
      bf16_t bv = (bf16_t)v;
      sA[s * 128 + h] = bv;
      sAT[h * 128 + s] = bv;
      acc += v * v;
    }
    sn1[s] = sqrtf(acc);
  } else {
    int t = tid - 128; float acc = 0.0f;
    const float* row = c2 + ((long)b * Sq + t) * (2 * Hq) + hoff;
    for (int h = 0; h < 128; ++h) {
      float v = row[h];
      bf16_t bv = (bf16_t)v;
      sB[t * 128 + h] = bv;
      sBT[h * 128 + t] = bv;
      acc += v * v;
    }
    sn2[t] = sqrtf(acc);
  }
  __syncthreads();

  // alpha[s,t] = cos(p1_s, p2_t) with safe_div on n1*n2; store both layouts
#pragma unroll
  for (int i = 0; i < 8; ++i) {
    int tile = wave * 8 + i;
    int mt = tile >> 3, nt = tile & 7;
    v8f acc = zero8();
#pragma unroll
    for (int k0 = 0; k0 < 128; k0 += 32)
      acc = wmma_bf16(load_a_bf16(sA + mt * 16 * 128 + k0, 128, lane),
                      load_b_tr(sB + (nt * 16) * 128 + k0, 128, lane), acc);
    int n = lane & 15, half = lane >> 4;
#pragma unroll
    for (int r = 0; r < 8; ++r) {
      int s = mt * 16 + r + 8 * half, t = nt * 16 + n;
      float deno = sn1[s] * sn2[t];
      bf16_t av = (bf16_t)(acc[r] / (deno > EPSV ? deno : EPSV));
      aBF[s * 128 + t] = av;
      aTBF[t * 128 + s] = av;
    }
  }
  __syncthreads();

  if (tid < 128) {
    int s = tid; float sum = 0.0f, mx = -3.4e38f; int mi = 0;
    for (int t = 0; t < 128; ++t) {
      float a = (float)aBF[s * 128 + t];
      sum += a;
      if (a > mx) { mx = a; mi = t; }
    }
    rs1[s] = sum; idx1[s] = mi;
  } else {
    int t = tid - 128; float sum = 0.0f, mx = -3.4e38f; int mi = 0;
    for (int s = 0; s < 128; ++s) {
      float a = (float)aTBF[t * 128 + s];
      sum += a;
      if (a > mx) { mx = a; mi = s; }
    }
    rs2[t] = sum; idx2[t] = mi;
  }
  __syncthreads();

  // h_mat1 = alpha @ p2 (M=s,K=t,N=h) ; h_mat2 = alpha^T @ p1 (M=t,K=s,N=h)
#pragma unroll
  for (int i = 0; i < 16; ++i) {
    int tile = wave * 16 + i;        // 0..127
    int which = tile >> 6;           // 0 -> h_mat1, 1 -> h_mat2
    int tt = tile & 63;
    int mt = tt >> 3, nt = tt & 7;
    v8f acc = zero8();
    if (which == 0) {
#pragma unroll
      for (int k0 = 0; k0 < 128; k0 += 32)
        acc = wmma_bf16(load_a_bf16(aBF + mt * 16 * 128 + k0, 128, lane),
                        load_b_tr(sBT + (nt * 16) * 128 + k0, 128, lane), acc);
    } else {
#pragma unroll
      for (int k0 = 0; k0 < 128; k0 += 32)
        acc = wmma_bf16(load_a_bf16(aTBF + mt * 16 * 128 + k0, 128, lane),
                        load_b_tr(sAT + (nt * 16) * 128 + k0, 128, lane), acc);
    }
    int n = lane & 15, half = lane >> 4;
#pragma unroll
    for (int r = 0; r < 8; ++r) {
      int m = mt * 16 + r + 8 * half;
      int hcol = nt * 16 + n;
      if (which == 0)
        r1[((long)b * Sq + m) * Hq + hcol] = acc[r] / rs1[m];
      else
        r2[((long)b * Sq + m) * Hq + hcol] = acc[r] / rs2[m];
    }
  }

  // out_mat gathers (argmax rows)
  if (tid < 128) {
    int s = tid, j = idx1[s];
    const float* row = c2 + ((long)b * Sq + j) * (2 * Hq) + hoff;
    for (int h = 0; h < 128; ++h) o1[((long)b * Sq + s) * Hq + h] = row[h];
  } else {
    int t = tid - 128, j = idx2[t];
    const float* row = c1 + ((long)b * Sq + j) * (2 * Hq) + hoff;
    for (int h = 0; h < 128; ++h) o2[((long)b * Sq + t) * Hq + h] = row[h];
  }
}

// ---------------------------------------------------------------------------
// Finisher: per (b,s) block computes full-matching + _wcos of attentive
// outputs for one side, writing aggr channels 0..39 and 80..159.
// ---------------------------------------------------------------------------
__global__ void k_finish(const float* __restrict__ cs,  // self context (B,S,256)
                         const float* __restrict__ co,  // other context (B,S,256)
                         const float* __restrict__ resf, const float* __restrict__ resb,
                         const float* __restrict__ omf, const float* __restrict__ omb,
                         const float* __restrict__ w1, const float* __restrict__ w2,
                         const float* __restrict__ w5, const float* __restrict__ w6,
                         const float* __restrict__ w7, const float* __restrict__ w8,
                         float* __restrict__ aggr) {
  long bs = blockIdx.x;
  long b = bs / Sq;
  __shared__ float p1r[256], p2l[256], rf[128], rb[128], ofv[128], obv[128];
  int tid = threadIdx.x; // 128 threads
  for (int i = tid; i < 256; i += 128) {
    p1r[i] = cs[bs * 256 + i];
    p2l[i] = co[(b * Sq + (Sq - 1)) * 256 + i];
  }
  rf[tid] = resf[bs * Hq + tid];
  rb[tid] = resb[bs * Hq + tid];
  ofv[tid] = omf[bs * Hq + tid];
  obv[tid] = omb[bs * Hq + tid];
  __syncthreads();
  if (tid < 6 * Lq) {
    int g = tid / Lq, l = tid % Lq;
    const float* a = p1r; const float* bv = p2l; const float* w = w1; int off = 0;
    switch (g) {
      case 0: a = p1r;       bv = p2l;       w = w1; off = 0;   break;
      case 1: a = p1r + 128; bv = p2l + 128; w = w2; off = 20;  break;
      case 2: a = rf;        bv = p1r;       w = w5; off = 80;  break;
      case 3: a = rb;        bv = p1r + 128; w = w6; off = 100; break;
      case 4: a = ofv;       bv = p1r;       w = w7; off = 120; break;
      default: a = obv;      bv = p1r + 128; w = w8; off = 140; break;
    }
    float dot = 0.0f, na = 0.0f, nb = 0.0f;
    for (int h = 0; h < 128; ++h) {
      float wv = w[h * Lq + l];
      float x = a[h] * wv, y = bv[h] * wv;
      dot += x * y; na += x * x; nb += y * y;
    }
    aggr[bs * (8 * Lq) + off + l] =
        dot / (fmaxf(sqrtf(na), EPSV) * fmaxf(sqrtf(nb), EPSV));
  }
}

// ---------------------------------------------------------------------------
// Head MLP: out = relu(cat @ pW1^T + pb1) @ pW2^T + pb2
// ---------------------------------------------------------------------------
__global__ void k_head(const float* __restrict__ hsPf, const float* __restrict__ hsPb,
                       const float* __restrict__ hsHf, const float* __restrict__ hsHb,
                       const float* __restrict__ pW1, const float* __restrict__ pb1,
                       const float* __restrict__ pW2, const float* __restrict__ pb2,
                       float* __restrict__ out) {
  __shared__ float cat[512];
  __shared__ float hid[256];
  long b = blockIdx.x;
  int tid = threadIdx.x; // 256
  long last = (b * Sq + (Sq - 1)) * Hq;
  if (tid < 128) {
    cat[tid] = hsPf[last + tid];
    cat[256 + tid] = hsHf[last + tid];
  } else {
    int h = tid - 128;
    cat[128 + h] = hsPb[last + h];
    cat[384 + h] = hsHb[last + h];
  }
  __syncthreads();
  float acc = pb1[tid];
  for (int k = 0; k < 512; ++k) acc += cat[k] * pW1[(long)tid * 512 + k];
  hid[tid] = fmaxf(acc, 0.0f);
  __syncthreads();
  if (tid < Cq) {
    float o = pb2[tid];
    for (int k = 0; k < 256; ++k) o += hid[k] * pW2[(long)tid * 256 + k];
    out[b * Cq + tid] = o;
  }
}

// ---------------------------------------------------------------------------
// Host-side orchestration
// ---------------------------------------------------------------------------
extern "C" void kernel_launch(void* const* d_in, const int* in_sizes, int n_in,
                              void* d_out, int out_size, void* d_ws, size_t ws_size,
                              hipStream_t stream) {
  (void)in_sizes; (void)n_in; (void)out_size; (void)ws_size;

  const float* emb       = (const float*)d_in[0];
  const float* ctx_wih_f = (const float*)d_in[1];
  const float* ctx_whh_f = (const float*)d_in[2];
  const float* ctx_b_f   = (const float*)d_in[3];
  const float* ctx_wih_b = (const float*)d_in[4];
  const float* ctx_whh_b = (const float*)d_in[5];
  const float* ctx_b_b   = (const float*)d_in[6];
  const float* agg_wih_f = (const float*)d_in[7];
  const float* agg_whh_f = (const float*)d_in[8];
  const float* agg_b_f   = (const float*)d_in[9];
  const float* agg_wih_b = (const float*)d_in[10];
  const float* agg_whh_b = (const float*)d_in[11];
  const float* agg_b_b   = (const float*)d_in[12];
  const float* pW1 = (const float*)d_in[13];
  const float* pb1 = (const float*)d_in[14];
  const float* pW2 = (const float*)d_in[15];
  const float* pb2 = (const float*)d_in[16];
  const int* pTok = (const int*)d_in[17];
  const int* hTok = (const int*)d_in[18];
  const float* wv[8];
  for (int i = 0; i < 8; ++i) wv[i] = (const float*)d_in[19 + i];
  float* out = (float*)d_out;

  // workspace slab allocation (256-byte aligned)
  char* ws = (char*)d_ws;
  size_t off = 0;
  auto take = [&](size_t bytes) -> char* {
    char* p = ws + off;
    off += (bytes + 255) & ~(size_t)255;
    return p;
  };
  const long BS = (long)Bq * Sq;
  bf16_t* xp_bf = (bf16_t*)take(BS * Dp * sizeof(bf16_t));
  bf16_t* xh_bf = (bf16_t*)take(BS * Dp * sizeof(bf16_t));
  bf16_t* wihCF = (bf16_t*)take((size_t)Dp * G4 * sizeof(bf16_t));
  bf16_t* wihCB = (bf16_t*)take((size_t)Dp * G4 * sizeof(bf16_t));
  bf16_t* wihAF = (bf16_t*)take((size_t)160 * G4 * sizeof(bf16_t));
  bf16_t* wihAB = (bf16_t*)take((size_t)160 * G4 * sizeof(bf16_t));
  bf16_t* whhCF = (bf16_t*)take((size_t)Hq * G4 * sizeof(bf16_t));
  bf16_t* whhCB = (bf16_t*)take((size_t)Hq * G4 * sizeof(bf16_t));
  bf16_t* whhAF = (bf16_t*)take((size_t)Hq * G4 * sizeof(bf16_t));
  bf16_t* whhAB = (bf16_t*)take((size_t)Hq * G4 * sizeof(bf16_t));
  float* xg0 = (float*)take(BS * G4 * sizeof(float));
  float* xg1 = (float*)take(BS * G4 * sizeof(float));
  float* xg2 = (float*)take(BS * G4 * sizeof(float));
  float* xg3 = (float*)take(BS * G4 * sizeof(float));
  float* c1 = (float*)take(BS * 2 * Hq * sizeof(float));
  float* c2 = (float*)take(BS * 2 * Hq * sizeof(float));
  float* aggr1 = (float*)take(BS * 8 * Lq * sizeof(float));
  float* aggr2 = (float*)take(BS * 8 * Lq * sizeof(float));
  bf16_t* aggr1_bf = (bf16_t*)take(BS * 8 * Lq * sizeof(bf16_t));
  bf16_t* aggr2_bf = (bf16_t*)take(BS * 8 * Lq * sizeof(bf16_t));
  float* res1 = (float*)take(2 * BS * Hq * sizeof(float)); // [dir][B,S,H]
  float* res2 = (float*)take(2 * BS * Hq * sizeof(float));
  float* om1  = (float*)take(2 * BS * Hq * sizeof(float));
  float* om2  = (float*)take(2 * BS * Hq * sizeof(float));
  float* hsPf = (float*)take(BS * Hq * sizeof(float));
  float* hsPb = (float*)take(BS * Hq * sizeof(float));
  float* hsHf = (float*)take(BS * Hq * sizeof(float));
  float* hsHb = (float*)take(BS * Hq * sizeof(float));

  // --- weight packing into WMMA-B fragment order ----------------------------
  {
    int n, g;
    n = Dp * G4; g = (n + 255) / 256;
    k_wpack<<<g, 256, 0, stream>>>(ctx_wih_f, wihCF, G4, Dq, Dp);
    k_wpack<<<g, 256, 0, stream>>>(ctx_wih_b, wihCB, G4, Dq, Dp);
    n = 160 * G4; g = (n + 255) / 256;
    k_wpack<<<g, 256, 0, stream>>>(agg_wih_f, wihAF, G4, 160, 160);
    k_wpack<<<g, 256, 0, stream>>>(agg_wih_b, wihAB, G4, 160, 160);
    n = Hq * G4; g = (n + 255) / 256;
    k_wpack<<<g, 256, 0, stream>>>(ctx_whh_f, whhCF, G4, Hq, Hq);
    k_wpack<<<g, 256, 0, stream>>>(ctx_whh_b, whhCB, G4, Hq, Hq);
    k_wpack<<<g, 256, 0, stream>>>(agg_whh_f, whhAF, G4, Hq, Hq);
    k_wpack<<<g, 256, 0, stream>>>(agg_whh_b, whhAB, G4, Hq, Hq);
  }

  // --- embedding gather ------------------------------------------------------
  k_embed<<<(int)BS, 256, 0, stream>>>(emb, pTok, xp_bf);
  k_embed<<<(int)BS, 256, 0, stream>>>(emb, hTok, xh_bf);

  // --- context input projections: xg = x @ wih^T + b ------------------------
  {
    int tiles = (int)(BS / 16) * (G4 / 16);
    int blocks = tiles / 8;
    k_gemm<<<blocks, 256, 0, stream>>>(xp_bf, wihCF, ctx_b_f, xg0, (int)BS, G4, Dp);
    k_gemm<<<blocks, 256, 0, stream>>>(xp_bf, wihCB, ctx_b_b, xg1, (int)BS, G4, Dp);
    k_gemm<<<blocks, 256, 0, stream>>>(xh_bf, wihCF, ctx_b_f, xg2, (int)BS, G4, Dp);
    k_gemm<<<blocks, 256, 0, stream>>>(xh_bf, wihCB, ctx_b_b, xg3, (int)BS, G4, Dp);
  }

  // --- context BiLSTM scans (persistent, whh in LDS) ------------------------
  const size_t lstm_smem = 128 * 512 * 2 + 16 * 512 * 4 + 16 * 128 * 4 + 16 * 128 * 2;
  k_lstm<<<Bq / 16, 256, lstm_smem, stream>>>(xg0, whhCF, c1, 2 * Hq, 0, 0);
  k_lstm<<<Bq / 16, 256, lstm_smem, stream>>>(xg1, whhCB, c1, 2 * Hq, Hq, 1);
  k_lstm<<<Bq / 16, 256, lstm_smem, stream>>>(xg2, whhCF, c2, 2 * Hq, 0, 0);
  k_lstm<<<Bq / 16, 256, lstm_smem, stream>>>(xg3, whhCB, c2, 2 * Hq, Hq, 1);

  // --- maxpool matching (fused cosine + max reduction) ----------------------
  const size_t mp_smem = 2 * 128 * 128 * sizeof(bf16_t) + 4 * 128 * 4 + 128 * 4;
  k_maxpool<<<Bq * Lq * 2, 256, mp_smem, stream>>>(c1, c2, wv[2], wv[3], aggr1, aggr2);

  // --- attentive matching core ----------------------------------------------
  const size_t at_smem = 6 * 128 * 128 * sizeof(bf16_t) + 6 * 128 * 4;
  k_attentive<<<Bq * 2, 256, at_smem, stream>>>(c1, c2, res1, res2, om1, om2);

  // --- finishers (full match + weighted cosines) ----------------------------
  const long BSH = BS * Hq;
  k_finish<<<(int)BS, 128, 0, stream>>>(c1, c2, res1, res1 + BSH, om1, om1 + BSH,
                                        wv[0], wv[1], wv[4], wv[5], wv[6], wv[7],
                                        aggr1);
  k_finish<<<(int)BS, 128, 0, stream>>>(c2, c1, res2, res2 + BSH, om2, om2 + BSH,
                                        wv[0], wv[1], wv[4], wv[5], wv[6], wv[7],
                                        aggr2);

  // --- aggregation BiLSTM ----------------------------------------------------
  {
    int n = (int)(BS * 8 * Lq);
    k_cvt_bf16<<<(n + 255) / 256, 256, 0, stream>>>(aggr1, aggr1_bf, n);
    k_cvt_bf16<<<(n + 255) / 256, 256, 0, stream>>>(aggr2, aggr2_bf, n);
    int tiles = (int)(BS / 16) * (G4 / 16);
    int blocks = tiles / 8;
    k_gemm<<<blocks, 256, 0, stream>>>(aggr1_bf, wihAF, agg_b_f, xg0, (int)BS, G4, 160);
    k_gemm<<<blocks, 256, 0, stream>>>(aggr1_bf, wihAB, agg_b_b, xg1, (int)BS, G4, 160);
    k_gemm<<<blocks, 256, 0, stream>>>(aggr2_bf, wihAF, agg_b_f, xg2, (int)BS, G4, 160);
    k_gemm<<<blocks, 256, 0, stream>>>(aggr2_bf, wihAB, agg_b_b, xg3, (int)BS, G4, 160);
  }
  k_lstm<<<Bq / 16, 256, lstm_smem, stream>>>(xg0, whhAF, hsPf, Hq, 0, 0);
  k_lstm<<<Bq / 16, 256, lstm_smem, stream>>>(xg1, whhAB, hsPb, Hq, 0, 1);
  k_lstm<<<Bq / 16, 256, lstm_smem, stream>>>(xg2, whhAF, hsHf, Hq, 0, 0);
  k_lstm<<<Bq / 16, 256, lstm_smem, stream>>>(xg3, whhAB, hsHb, Hq, 0, 1);

  // --- head ------------------------------------------------------------------
  k_head<<<Bq, 256, 0, stream>>>(hsPf, hsPb, hsHf, hsHb, pW1, pb1, pW2, pb2, out);
}
